// in_order_constituent_parser_11630771438029
// MI455X (gfx1250) — matrix operand order repaired and checked by reading, not compiled, not run on hardware
//
#include <hip/hip_runtime.h>
#include <hip/hip_bf16.h>
#include <math.h>

// Problem dims (fixed by reference setup_inputs)
#define T_DIM  4096
#define S_DIM  2048
#define H_DIM  1024
#define D_DIM  1024
#define F_DIM  2048
#define A_DIM  2048
#define L_DIM  2
#define FOURH  4096
#define CAT_DIM 3072
#define NEG_BIG 10000000000.0f

typedef __attribute__((ext_vector_type(16))) __bf16    v16bf;
typedef __attribute__((ext_vector_type(8)))  float     v8f;
typedef __attribute__((ext_vector_type(4)))  unsigned  u32x4;

// ---------------- helpers ----------------
__device__ __forceinline__ unsigned short f2bf(float f) {
  unsigned u = __float_as_uint(f);
  unsigned r = (u + 0x7fffu + ((u >> 16) & 1u)) >> 16;  // RNE
  return (unsigned short)r;
}
__device__ __forceinline__ float bf2f_lo(unsigned u) {  // low bf16 of packed pair
  return __uint_as_float(u << 16);
}
__device__ __forceinline__ float bf2f_hi(unsigned u) {  // high bf16 of packed pair
  return __uint_as_float(u & 0xffff0000u);
}
__device__ __forceinline__ float sigmoidf_(float x) { return 1.0f / (1.0f + expf(-x)); }

__device__ __forceinline__ float block_reduce_max(float v, float* red) {
  int tid = threadIdx.x;
  red[tid] = v; __syncthreads();
  for (int s = 128; s > 0; s >>= 1) {
    if (tid < s) red[tid] = fmaxf(red[tid], red[tid + s]);
    __syncthreads();
  }
  float r = red[0]; __syncthreads();
  return r;
}
__device__ __forceinline__ float block_reduce_sum(float v, float* red) {
  int tid = threadIdx.x;
  red[tid] = v; __syncthreads();
  for (int s = 128; s > 0; s >>= 1) {
    if (tid < s) red[tid] = red[tid] + red[tid + s];
    __syncthreads();
  }
  float r = red[0]; __syncthreads();
  return r;
}

// ---------------- small prep kernels ----------------
__global__ void convert_f32_bf16_kernel(const float* __restrict__ src,
                                        unsigned short* __restrict__ dst, size_t n) {
  size_t stride = (size_t)gridDim.x * blockDim.x;
  for (size_t i = (size_t)blockIdx.x * blockDim.x + threadIdx.x; i < n; i += stride)
    dst[i] = f2bf(src[i]);
}

__global__ void transpose_convert_kernel(const float* __restrict__ enc,
                                         unsigned short* __restrict__ encT) {
  size_t n = (size_t)S_DIM * H_DIM;
  size_t stride = (size_t)gridDim.x * blockDim.x;
  for (size_t i = (size_t)blockIdx.x * blockDim.x + threadIdx.x; i < n; i += stride) {
    size_t s = i / H_DIM, h = i - s * H_DIM;
    encT[h * S_DIM + s] = f2bf(enc[i]);
  }
}

__global__ void bias_sum_kernel(const float* __restrict__ a, const float* __restrict__ b,
                                float* __restrict__ o, int n) {
  int i = blockIdx.x * 256 + threadIdx.x;
  if (i < n) o[i] = a[i] + b[i];
}

// action_embeddings[t] = (t==0) ? 0 : embeds[actions[t-1]]; bf16 to GEMM-A buffer
// and straight into the concat (feat_in) column block [2H, 2H+D).
__global__ __launch_bounds__(256) void gather_embed_kernel(
    const int* __restrict__ actions, const float* __restrict__ embeds,
    unsigned short* __restrict__ embbf, unsigned short* __restrict__ featin) {
  int t = blockIdx.x;
  int row = (t == 0) ? -1 : actions[t - 1];
  for (int i = threadIdx.x; i < D_DIM; i += 256) {
    float v = (row < 0) ? 0.0f : embeds[(size_t)row * D_DIM + i];
    unsigned short b = f2bf(v);
    embbf[(size_t)t * D_DIM + i] = b;
    featin[(size_t)t * CAT_DIM + 2 * H_DIM + i] = b;
  }
}

// ---------------- WMMA GEMM: C[M,N] = A[M,K] * B[N,K]^T (+bias) (tanh?) ----------------
// A,B row-major bf16 (K contiguous). Block tile 128x128, 8 waves, wave tile 32x64.
// LDS rows padded to 20 dwords (80 B): keeps every fragment run 16B-aligned so all
// fragment gathers are ds_load_b128, and 20 mod 64 banks rotates rows conflict-free.
template <int OUT_BF16, int DO_TANH>
__global__ __launch_bounds__(256) void gemm_bf16_kernel(
    const unsigned short* __restrict__ A, const unsigned short* __restrict__ B,
    void* __restrict__ Cout, const float* __restrict__ bias,
    int M, int N, int K, int ldc) {
  __shared__ __align__(16) unsigned sA[128][20];  // cols 0..15 hold 16 bf16-pairs
  __shared__ __align__(16) unsigned sB[128][20];
  const int tid  = threadIdx.x;
  const int lane = tid & 31;
  const int w    = tid >> 5;
  const int waveM = w >> 1;   // 0..3
  const int waveN = w & 1;    // 0..1
  const int hi   = lane >> 4; // half-wave select
  const int l15  = lane & 15;
  const int bM = blockIdx.y * 128;
  const int bN = blockIdx.x * 128;
  const int Kp4 = K >> 3;     // u32x4 chunks per row
  const u32x4* Ag = (const u32x4*)A;
  const u32x4* Bg = (const u32x4*)B;
  (void)M; (void)N;

  v8f acc[2][4];
  const v8f vzero = {0.f, 0.f, 0.f, 0.f, 0.f, 0.f, 0.f, 0.f};
#pragma unroll
  for (int i = 0; i < 2; ++i)
#pragma unroll
    for (int j = 0; j < 4; ++j) acc[i][j] = vzero;

  union FragBF { u32x4 q[2]; v16bf v; };

  const int ktiles = K >> 5;
  for (int kt = 0; kt < ktiles; ++kt) {
    // 512 16-byte chunks per 128x32 tile; 2 per thread, b128 load + b128 store.
#pragma unroll
    for (int i = 0; i < 2; ++i) {
      int chunk = tid + i * 256;      // 0..511
      int r  = chunk >> 2;            // 0..127
      int cg = chunk & 3;             // 16-byte group within the 32-K slab
      *(u32x4*)&sA[r][cg * 4] = Ag[(size_t)(bM + r) * Kp4 + (size_t)kt * 4 + cg];
      *(u32x4*)&sB[r][cg * 4] = Bg[(size_t)(bN + r) * Kp4 + (size_t)kt * 4 + cg];
    }
    __syncthreads();

    // ISA 16-bit A 16x32 layout: lane<16 -> pairs {0..3}=K0..7, {4..7}=K16..23; lane>=16 +4 pairs.
    // ISA 16-bit B 32x16 layout: same K order as A: lane<16 -> pairs {0..3}=K0..7, {4..7}=K16..23; lane>=16 +4 pairs.
    FragBF af[2], bfm[4];
#pragma unroll
    for (int mi = 0; mi < 2; ++mi) {
      int row = waveM * 32 + mi * 16 + l15;
      af[mi].q[0] = *(const u32x4*)&sA[row][hi * 4];
      af[mi].q[1] = *(const u32x4*)&sA[row][hi * 4 + 8];
    }
#pragma unroll
    for (int ni = 0; ni < 4; ++ni) {
      int col = waveN * 64 + ni * 16 + l15;
      bfm[ni].q[0] = *(const u32x4*)&sB[col][hi * 4];
      bfm[ni].q[1] = *(const u32x4*)&sB[col][hi * 4 + 8];
    }
#pragma unroll
    for (int mi = 0; mi < 2; ++mi)
#pragma unroll
      for (int ni = 0; ni < 4; ++ni)
        acc[mi][ni] = __builtin_amdgcn_wmma_f32_16x16x32_bf16(
            false, af[mi].v, false, bfm[ni].v, (short)0, acc[mi][ni], false, false);
    __syncthreads();
  }

  // Epilogue. C/D layout: lane = N (0..15, both halves), VGPR r -> M row, lanes>=16 -> M+8.
#pragma unroll
  for (int mi = 0; mi < 2; ++mi) {
    int rbase = bM + waveM * 32 + mi * 16 + hi * 8;
#pragma unroll
    for (int ni = 0; ni < 4; ++ni) {
      int cg = bN + waveN * 64 + ni * 16 + l15;
      float bv = bias ? bias[cg] : 0.0f;
#pragma unroll
      for (int r = 0; r < 8; ++r) {
        float val = acc[mi][ni][r] + bv;
        if (DO_TANH) val = tanhf(val);
        size_t idx = (size_t)(rbase + r) * (size_t)ldc + (size_t)cg;
        if (OUT_BF16) ((unsigned short*)Cout)[idx] = f2bf(val);
        else          ((float*)Cout)[idx] = val;
      }
    }
  }
}

// ---------------- persistent LSTM recurrence ----------------
// 64 WGs x 256 threads. WG owns 16 h-elements; its 64 w_hh gate rows live in LDS as bf16
// (128 KB -- only possible with CDNA5's 320 KB WGP LDS). One grid barrier per time step.
#define RNWG 64
#define LSTM_LDS (64 * H_DIM * 2 + (H_DIM + 64) * 4)

__global__ __launch_bounds__(256) void lstm_rec_kernel(
    const float* __restrict__ whh,       // [4H, H] this layer
    const float* __restrict__ xg,        // [T, 4H] precomputed x@w_ih^T + bias
    unsigned short* __restrict__ outbf,  // [T, H] bf16 output sequence
    float* __restrict__ hglob,           // [H] zeroed before launch
    unsigned* __restrict__ bar) {        // zeroed before launch
  extern __shared__ __align__(16) char smem[];
  unsigned short* wl = (unsigned short*)smem;                                  // [64][H] bf16
  float* hbuf = (float*)(smem + (size_t)64 * H_DIM * sizeof(unsigned short));  // [H]
  float* gdot = hbuf + H_DIM;                                                  // [64]

  const int tid = threadIdx.x;
  const int wg  = blockIdx.x;
  const int CH  = H_DIM / RNWG;  // 16 h-elements per WG
  const int base = wg * CH;

  // Stage this WG's 64 gate rows of w_hh into LDS as bf16, once.
  for (int idx = tid; idx < 64 * H_DIM; idx += 256) {
    int lr = idx / H_DIM;
    int k  = idx - lr * H_DIM;
    int g  = lr / CH;
    int j  = lr - g * CH;
    wl[idx] = f2bf(whh[(size_t)(g * H_DIM + base + j) * H_DIM + k]);
  }
  float c = 0.0f;  // cell state, valid for tid < CH
  __syncthreads();

  const int r = tid >> 2;  // local gate row 0..63
  const int p = tid & 3;   // quarter of the dot product
  // Wide LDS pointers: weights as 8-bf16 (16 B) chunks, h as float4.
  const u32x4*  wq = ((const u32x4*)wl) + r * (H_DIM / 8) + p * (H_DIM / 32);
  const float4* hq = (const float4*)(hbuf + p * (H_DIM / 4));

  unsigned epoch = 0;
  for (int t = 0; t < T_DIM; ++t) {
    for (int k = tid; k < H_DIM; k += 256) hbuf[k] = hglob[k];
    __syncthreads();

    float s = 0.0f;
#pragma unroll 4
    for (int k = 0; k < H_DIM / 32; ++k) {  // 32 iters x 8 MACs
      u32x4  wv = wq[k];
      float4 h0 = hq[2 * k];
      float4 h1 = hq[2 * k + 1];
      s += bf2f_lo(wv[0]) * h0.x + bf2f_hi(wv[0]) * h0.y;
      s += bf2f_lo(wv[1]) * h0.z + bf2f_hi(wv[1]) * h0.w;
      s += bf2f_lo(wv[2]) * h1.x + bf2f_hi(wv[2]) * h1.y;
      s += bf2f_lo(wv[3]) * h1.z + bf2f_hi(wv[3]) * h1.w;
    }
    s += __shfl_down(s, 1, 4);
    s += __shfl_down(s, 2, 4);
    if (p == 0) gdot[r] = s;
    __syncthreads();

    if (tid < CH) {
      int j = tid;
      const float* xr = xg + (size_t)t * FOURH;
      float gi = xr[0 * H_DIM + base + j] + gdot[0 * CH + j];
      float gf = xr[1 * H_DIM + base + j] + gdot[1 * CH + j];
      float gg = xr[2 * H_DIM + base + j] + gdot[2 * CH + j];
      float go = xr[3 * H_DIM + base + j] + gdot[3 * CH + j];
      gi = sigmoidf_(gi); gf = sigmoidf_(gf); gg = tanhf(gg); go = sigmoidf_(go);
      c = gf * c + gi * gg;
      float h = go * tanhf(c);
      hglob[base + j] = h;
      outbf[(size_t)t * H_DIM + base + j] = f2bf(h);
    }
    __threadfence();
    __syncthreads();
    ++epoch;
    if (tid == 0) {
      __hip_atomic_fetch_add(bar, 1u, __ATOMIC_RELEASE, __HIP_MEMORY_SCOPE_AGENT);
      while (__hip_atomic_load(bar, __ATOMIC_ACQUIRE, __HIP_MEMORY_SCOPE_AGENT) <
             epoch * (unsigned)RNWG) {
        __builtin_amdgcn_s_sleep(2);
      }
    }
    __syncthreads();
  }
}

// ---------------- dual masked softmax over S ----------------
__global__ __launch_bounds__(256) void dual_softmax_kernel(
    const float* __restrict__ scores, const float* __restrict__ smask,
    const float* __restrict__ bmask, unsigned short* __restrict__ wS,
    unsigned short* __restrict__ wB) {
  __shared__ float red[256];
  const int t = blockIdx.x, tid = threadIdx.x;
  const size_t rowo = (size_t)t * S_DIM;
  float vs[8], vb[8];
#pragma unroll
  for (int i = 0; i < 8; ++i) {
    int col = tid + i * 256;
    float sc = scores[rowo + col];
    vs[i] = sc + (smask[rowo + col] - 1.0f) * NEG_BIG;
    vb[i] = sc + (bmask[rowo + col] - 1.0f) * NEG_BIG;
  }
  float ms = -INFINITY, mb = -INFINITY;
#pragma unroll
  for (int i = 0; i < 8; ++i) { ms = fmaxf(ms, vs[i]); mb = fmaxf(mb, vb[i]); }
  ms = block_reduce_max(ms, red);
  mb = block_reduce_max(mb, red);
  float es = 0.0f, eb = 0.0f;
#pragma unroll
  for (int i = 0; i < 8; ++i) {
    vs[i] = expf(vs[i] - ms); es += vs[i];
    vb[i] = expf(vb[i] - mb); eb += vb[i];
  }
  es = block_reduce_sum(es, red);
  eb = block_reduce_sum(eb, red);
  float is_ = 1.0f / es, ib_ = 1.0f / eb;
#pragma unroll
  for (int i = 0; i < 8; ++i) {
    int col = tid + i * 256;
    wS[rowo + col] = f2bf(vs[i] * is_);
    wB[rowo + col] = f2bf(vb[i] * ib_);
  }
}

// ---------------- masked log-softmax over A (in-place on d_out) ----------------
__global__ __launch_bounds__(256) void log_softmax_kernel(
    float* __restrict__ dist, const float* __restrict__ mask) {
  __shared__ float red[256];
  const int t = blockIdx.x, tid = threadIdx.x;
  const size_t rowo = (size_t)t * A_DIM;
  float v[8];
#pragma unroll
  for (int i = 0; i < 8; ++i) {
    int col = tid + i * 256;
    v[i] = dist[rowo + col] + (mask[rowo + col] - 1.0f) * NEG_BIG;
  }
  float m = -INFINITY;
#pragma unroll
  for (int i = 0; i < 8; ++i) m = fmaxf(m, v[i]);
  m = block_reduce_max(m, red);
  float e = 0.0f;
#pragma unroll
  for (int i = 0; i < 8; ++i) e += expf(v[i] - m);
  e = block_reduce_sum(e, red);
  float lse = m + logf(e);
#pragma unroll
  for (int i = 0; i < 8; ++i) {
    int col = tid + i * 256;
    dist[rowo + col] = v[i] - lse;
  }
}

// ---------------- launch ----------------
extern "C" void kernel_launch(void* const* d_in, const int* in_sizes, int n_in,
                              void* d_out, int out_size, void* d_ws, size_t ws_size,
                              hipStream_t stream) {
  (void)in_sizes; (void)n_in; (void)out_size; (void)ws_size;
  const int*   actions = (const int*)d_in[0];
  const float* masks   = (const float*)d_in[1];
  const float* smask   = (const float*)d_in[2];
  const float* bmask   = (const float*)d_in[3];
  const float* enc     = (const float*)d_in[4];
  const float* embeds  = (const float*)d_in[5];
  const float* w_ih    = (const float*)d_in[6];
  const float* w_hh    = (const float*)d_in[7];
  const float* b_ih    = (const float*)d_in[8];
  const float* b_hh    = (const float*)d_in[9];
  const float* feat_W  = (const float*)d_in[10];
  const float* feat_b  = (const float*)d_in[11];
  const float* out_W   = (const float*)d_in[12];
  const float* out_b   = (const float*)d_in[13];
  float* out = (float*)d_out;

  // workspace bump allocator (256B-aligned regions)
  size_t off = 0;
  char* wsb = (char*)d_ws;
  auto alloc = [&](size_t bytes) -> void* {
    off = (off + 255) & ~(size_t)255;
    void* p = wsb + off;
    off += bytes;
    return p;
  };
  unsigned short* emb_bf    = (unsigned short*)alloc((size_t)T_DIM * D_DIM * 2);
  float*          xg        = (float*)alloc((size_t)T_DIM * FOURH * 4);
  unsigned short* out0_bf   = (unsigned short*)alloc((size_t)T_DIM * H_DIM * 2);
  unsigned short* out1_bf   = (unsigned short*)alloc((size_t)T_DIM * H_DIM * 2);
  float*          hglob     = (float*)alloc((size_t)H_DIM * 4);
  unsigned*       bar       = (unsigned*)alloc(256);
  float*          bsum      = (float*)alloc((size_t)L_DIM * FOURH * 4);
  unsigned short* wih_bf    = (unsigned short*)alloc((size_t)L_DIM * FOURH * D_DIM * 2);
  unsigned short* enc_bf    = (unsigned short*)alloc((size_t)S_DIM * H_DIM * 2);
  unsigned short* encT_bf   = (unsigned short*)alloc((size_t)H_DIM * S_DIM * 2);
  unsigned short* featW_bf  = (unsigned short*)alloc((size_t)F_DIM * CAT_DIM * 2);
  unsigned short* outW_bf   = (unsigned short*)alloc((size_t)A_DIM * F_DIM * 2);
  float*          scores    = (float*)alloc((size_t)T_DIM * S_DIM * 4);
  unsigned short* attnS_bf  = (unsigned short*)alloc((size_t)T_DIM * S_DIM * 2);
  unsigned short* attnB_bf  = (unsigned short*)alloc((size_t)T_DIM * S_DIM * 2);
  unsigned short* featin_bf = (unsigned short*)alloc((size_t)T_DIM * CAT_DIM * 2);
  unsigned short* feath_bf  = (unsigned short*)alloc((size_t)T_DIM * F_DIM * 2);

  // --- weight prep (fp32 -> bf16) ---
  convert_f32_bf16_kernel<<<2048, 256, 0, stream>>>(w_ih, wih_bf, (size_t)L_DIM * FOURH * D_DIM);
  convert_f32_bf16_kernel<<<2048, 256, 0, stream>>>(feat_W, featW_bf, (size_t)F_DIM * CAT_DIM);
  convert_f32_bf16_kernel<<<2048, 256, 0, stream>>>(out_W, outW_bf, (size_t)A_DIM * F_DIM);
  convert_f32_bf16_kernel<<<2048, 256, 0, stream>>>(enc, enc_bf, (size_t)S_DIM * H_DIM);
  transpose_convert_kernel<<<2048, 256, 0, stream>>>(enc, encT_bf);
  bias_sum_kernel<<<(L_DIM * FOURH + 255) / 256, 256, 0, stream>>>(b_ih, b_hh, bsum, L_DIM * FOURH);
  gather_embed_kernel<<<T_DIM, 256, 0, stream>>>(actions, embeds, emb_bf, featin_bf);

  // --- layer 0: xg = emb @ w_ih0^T + (b_ih0+b_hh0); then recurrence ---
  dim3 g4h(FOURH / 128, T_DIM / 128);
  gemm_bf16_kernel<0, 0><<<g4h, 256, 0, stream>>>(emb_bf, wih_bf, xg, bsum,
                                                  T_DIM, FOURH, D_DIM, FOURH);
  hipMemsetAsync(hglob, 0, (size_t)H_DIM * 4, stream);
  hipMemsetAsync(bar, 0, 256, stream);
  lstm_rec_kernel<<<RNWG, 256, LSTM_LDS, stream>>>(w_hh, xg, out0_bf, hglob, bar);

  // --- layer 1 ---
  gemm_bf16_kernel<0, 0><<<g4h, 256, 0, stream>>>(out0_bf, wih_bf + (size_t)FOURH * D_DIM,
                                                  xg, bsum + FOURH, T_DIM, FOURH, H_DIM, FOURH);
  hipMemsetAsync(hglob, 0, (size_t)H_DIM * 4, stream);
  hipMemsetAsync(bar, 0, 256, stream);
  lstm_rec_kernel<<<RNWG, 256, LSTM_LDS, stream>>>(w_hh + (size_t)FOURH * H_DIM, xg,
                                                   out1_bf, hglob, bar);

  // --- attention scores + dual masked softmax ---
  dim3 gsc(S_DIM / 128, T_DIM / 128);
  gemm_bf16_kernel<0, 0><<<gsc, 256, 0, stream>>>(out1_bf, enc_bf, scores, nullptr,
                                                  T_DIM, S_DIM, H_DIM, S_DIM);
  dual_softmax_kernel<<<T_DIM, 256, 0, stream>>>(scores, smask, bmask, attnS_bf, attnB_bf);

  // --- attn context vectors written straight into concat buffer (ldc = 3072) ---
  dim3 gah(H_DIM / 128, T_DIM / 128);
  gemm_bf16_kernel<1, 0><<<gah, 256, 0, stream>>>(attnS_bf, encT_bf, featin_bf, nullptr,
                                                  T_DIM, H_DIM, S_DIM, CAT_DIM);
  gemm_bf16_kernel<1, 0><<<gah, 256, 0, stream>>>(attnB_bf, encT_bf, featin_bf + H_DIM, nullptr,
                                                  T_DIM, H_DIM, S_DIM, CAT_DIM);

  // --- feature MLP (fused bias+tanh, bf16 out) ---
  dim3 gft(F_DIM / 128, T_DIM / 128);
  gemm_bf16_kernel<1, 1><<<gft, 256, 0, stream>>>(featin_bf, featW_bf, feath_bf, feat_b,
                                                  T_DIM, F_DIM, CAT_DIM, F_DIM);

  // --- output head into d_out, then masked log-softmax in place ---
  dim3 gds(A_DIM / 128, T_DIM / 128);
  gemm_bf16_kernel<0, 0><<<gds, 256, 0, stream>>>(feath_bf, outW_bf, out, out_b,
                                                  T_DIM, A_DIM, F_DIM, A_DIM);
  log_softmax_kernel<<<T_DIM, 256, 0, stream>>>(out, masks);
}
